// SimpleMambaClassifier_3848290697799
// MI455X (gfx1250) — matrix operand order repaired and checked by reading, not compiled, not run on hardware
//
#include <hip/hip_runtime.h>
#include <stdint.h>

// ---------------------------------------------------------------------------
// Model constants (from reference)
// ---------------------------------------------------------------------------
#define BB      16
#define HH      64
#define WW      64
#define LL      (HH * WW)          // 4096
#define BL      (BB * LL)          // 65536
#define DM      512                // d_model
#define DI      1024               // d_inner
#define DS      16                 // d_state
#define DC      4                  // d_conv
#define DTR     32                 // dt_rank
#define NL      4                  // layers
#define NC      10                 // classes

// ---------------------------------------------------------------------------
// WMMA vector types (gfx1250, wave32)
// ---------------------------------------------------------------------------
typedef __attribute__((ext_vector_type(16))) __bf16 v16bf;
typedef __attribute__((ext_vector_type(8)))  __bf16 v8bf;
typedef __attribute__((ext_vector_type(8)))  float  v8f;

enum { GEMM_F32 = 0, GEMM_BF16 = 1, GEMM_SP_BIAS_BF16 = 2 };

// ---------------------------------------------------------------------------
// CDNA5 async global->LDS copy (ASYNCcnt-tracked, no VGPR round trip)
// ---------------------------------------------------------------------------
__device__ __forceinline__ void async_copy_b128(const void* gptr, void* lptr)
{
    uint32_t laddr = (uint32_t)(size_t)lptr;          // AS3 offset = flat[31:0]
    uint64_t gaddr = (uint64_t)(size_t)gptr;
    asm volatile("global_load_async_to_lds_b128 %0, %1, off"
                 :: "v"(laddr), "v"(gaddr)
                 : "memory");
}

__device__ __forceinline__ void wait_async_zero()
{
#if __has_builtin(__builtin_amdgcn_s_wait_asynccnt)
    __builtin_amdgcn_s_wait_asynccnt(0);
#else
    asm volatile("s_wait_asynccnt 0x0" ::: "memory");
#endif
}

// ---------------------------------------------------------------------------
// Tiled bf16 WMMA GEMM:  C[M,N] = A[M,K] * W[N,K]^T   (W row-major N x K)
// Block tile 128x128, 8 waves (4M x 2N), wave tile 32x64 -> 8x v_wmma
// 16x16x32 per K-step. Double-buffered LDS filled by async-to-LDS copies;
// stride padded to 40 elems (80 B) for conflict-free b128 fragment reads.
// ---------------------------------------------------------------------------
#define TM   128
#define TN   128
#define TK   32
#define STR  40     // padded LDS row stride (elements); 80 B, 16-B aligned

__global__ __launch_bounds__(256)
void gemm_bf16_wmma(const __bf16* __restrict__ A, int lda,
                    const __bf16* __restrict__ W,
                    const float*  __restrict__ bias,
                    void* __restrict__ Cout, int ldc,
                    int M, int N, int K, int mode)
{
    __shared__ __attribute__((aligned(64))) __bf16 As[2][TM * STR];
    __shared__ __attribute__((aligned(64))) __bf16 Bs[2][TN * STR];

    const int tid   = threadIdx.x;
    const int lane  = tid & 31;
    const int wave  = tid >> 5;
    const int waveM = wave >> 1;    // 0..3  (32 rows each)
    const int waveN = wave & 1;     // 0..1  (64 cols each)
    const int mBase = blockIdx.x * TM;
    const int nBase = blockIdx.y * TN;

    v8f acc[2][4] = {};

    // staging indices: each thread moves 16 elems (2x b128) of A and of B
    const int sr = tid >> 1;          // 0..127 (row for A, n-row for B)
    const int sc = (tid & 1) * 16;    // 0,16

    // fragment indices (ISA 7.12.2)
    const int m16  = lane & 15;
    const int h8   = (lane >> 4) << 3;   // A frag K base {0,8}
    const int kb   = (lane >> 4) << 4;   // unused; B frag also uses h8 {0,8}
    const int half = lane >> 4;

    const bool bValid = (nBase + sr) < N;

    // zero-fill out-of-range B rows once (async copies never touch them)
    if (!bValid) {
        v8bf z = {};
        *(v8bf*)&Bs[0][sr * STR + sc]     = z;
        *(v8bf*)&Bs[0][sr * STR + sc + 8] = z;
        *(v8bf*)&Bs[1][sr * STR + sc]     = z;
        *(v8bf*)&Bs[1][sr * STR + sc + 8] = z;
    }

    // ---- async stage of one K-tile into buffer `buf` ----
    auto stage = [&](int buf, int kk) {
        const __bf16* ap = A + (size_t)(mBase + sr) * lda + kk + sc;
        async_copy_b128(ap,     &As[buf][sr * STR + sc]);
        async_copy_b128(ap + 8, &As[buf][sr * STR + sc + 8]);
        if (bValid) {
            const __bf16* wp = W + (size_t)(nBase + sr) * K + kk + sc;
            async_copy_b128(wp,     &Bs[buf][sr * STR + sc]);
            async_copy_b128(wp + 8, &Bs[buf][sr * STR + sc + 8]);
        }
    };

    // prologue: fill buffer 0
    stage(0, 0);
    wait_async_zero();
    __syncthreads();

    int cur = 0;
    for (int kk = 0; kk < K; kk += TK) {
        // prefill next K-tile into the other buffer while we compute
        if (kk + TK < K) {
            stage(cur ^ 1, kk + TK);
            // pull the tile after next toward L2 as well
            __builtin_prefetch(A + (size_t)(mBase + sr) * lda + kk + 2 * TK + sc, 0, 1);
            if (bValid)
                __builtin_prefetch(W + (size_t)(nBase + sr) * K + kk + 2 * TK + sc, 0, 1);
        }

        // ---- gather fragments from LDS (conflict-free b128 reads) ----
        v16bf afrag[2], bfrag[4];
#pragma unroll
        for (int i = 0; i < 2; ++i) {
            const __bf16* p = &As[cur][(waveM * 32 + i * 16 + m16) * STR + h8];
            v8bf lo = *(const v8bf*)p;          // K = h8 + 0..7
            v8bf hi = *(const v8bf*)(p + 16);   // K = h8 + 16..23
            afrag[i] = __builtin_shufflevector(lo, hi,
                0, 1, 2, 3, 4, 5, 6, 7, 8, 9, 10, 11, 12, 13, 14, 15);
        }
#pragma unroll
        for (int j = 0; j < 4; ++j) {
            const __bf16* p = &Bs[cur][(waveN * 64 + j * 16 + m16) * STR + h8];
            v8bf lo = *(const v8bf*)p;          // K = h8 + 0..7
            v8bf hi = *(const v8bf*)(p + 16);   // K = h8 + 16..23
            bfrag[j] = __builtin_shufflevector(lo, hi,
                0, 1, 2, 3, 4, 5, 6, 7, 8, 9, 10, 11, 12, 13, 14, 15);
        }

        // ---- 8x WMMA 16x16x32 bf16 -> f32 ----
#pragma unroll
        for (int i = 0; i < 2; ++i)
#pragma unroll
            for (int j = 0; j < 4; ++j)
                acc[i][j] = __builtin_amdgcn_wmma_f32_16x16x32_bf16(
                    false, afrag[i], false, bfrag[j],
                    (short)0, acc[i][j], false, false);

        wait_async_zero();     // next buffer's async fill has landed in LDS
        __syncthreads();
        cur ^= 1;
    }

    // ---- epilogue / store (C layout: VGPR r -> M = r + 8*half, N = m16) ----
#pragma unroll
    for (int i = 0; i < 2; ++i) {
#pragma unroll
        for (int j = 0; j < 4; ++j) {
            const int gn = nBase + waveN * 64 + j * 16 + m16;
            if (gn >= N) continue;
#pragma unroll
            for (int r = 0; r < 8; ++r) {
                const int gm = mBase + waveM * 32 + i * 16 + r + half * 8;
                float c = acc[i][j][r];
                size_t off = (size_t)gm * ldc + gn;
                if (mode == GEMM_F32) {
                    ((float*)Cout)[off] = c;
                } else if (mode == GEMM_BF16) {
                    ((__bf16*)Cout)[off] = (__bf16)c;
                } else { // softplus(c + bias)
                    float v  = c + bias[gn];
                    float sp = (v > 20.f) ? v : __logf(1.f + __expf(v));
                    ((__bf16*)Cout)[off] = (__bf16)sp;
                }
            }
        }
    }
}

// ---------------------------------------------------------------------------
// Elementwise helpers
// ---------------------------------------------------------------------------
__global__ void cvt_f32_bf16(const float* __restrict__ src,
                             __bf16* __restrict__ dst, int n)
{
    int i = blockIdx.x * blockDim.x + threadIdx.x;
    if (i < n) dst[i] = (__bf16)src[i];
}

// h[b,l,d] = sum_c x[b,c,l] * in_w[d,c] + in_b[d]; writes f32 + bf16
__global__ void embed_kernel(const float* __restrict__ x,
                             const float* __restrict__ in_w,
                             const float* __restrict__ in_b,
                             float* __restrict__ h,
                             __bf16* __restrict__ h_bf)
{
    int idx = blockIdx.x * blockDim.x + threadIdx.x;
    if (idx >= BL * DM) return;
    int dm = idx % DM;
    int bl = idx / DM;
    int b  = bl / LL;
    int l  = bl % LL;
    const float* xb = x + (size_t)b * 3 * LL;
    float v = in_b[dm];
    v += xb[0 * LL + l] * in_w[dm * 3 + 0];
    v += xb[1 * LL + l] * in_w[dm * 3 + 1];
    v += xb[2 * LL + l] * in_w[dm * 3 + 2];
    h[idx]    = v;
    h_bf[idx] = (__bf16)v;
}

// depthwise causal conv (width 4) over L + SiLU; reads xc part of xz (bf16)
__global__ void conv_silu_kernel(const __bf16* __restrict__ xz_bf,
                                 const float* __restrict__ cw,   // [DI,4]
                                 const float* __restrict__ cb,   // [DI]
                                 __bf16* __restrict__ u_bf)
{
    int idx = blockIdx.x * blockDim.x + threadIdx.x;
    if (idx >= BL * DI) return;
    int d  = idx % DI;
    int bl = idx / DI;
    int b  = bl / LL;
    int t  = bl % LL;
    float acc = cb[d];
#pragma unroll
    for (int j = 0; j < DC; ++j) {
        int tt = t - (DC - 1) + j;
        if (tt >= 0) {
            size_t row = (size_t)(b * LL + tt);
            acc += (float)xz_bf[row * (2 * DI) + d] * cw[d * DC + j];
        }
    }
    float s = acc / (1.f + __expf(-acc));   // silu
    u_bf[(size_t)bl * DI + d] = (__bf16)s;
}

// bf16 copy of first DTR columns of dbl (f32, ld 64) for the dt GEMM
__global__ void extract_dtA_kernel(const float* __restrict__ dbl,
                                   __bf16* __restrict__ dtA_bf)
{
    int idx = blockIdx.x * blockDim.x + threadIdx.x;
    if (idx >= BL * DTR) return;
    int r = idx % DTR;
    int m = idx / DTR;
    dtA_bf[idx] = (__bf16)dbl[(size_t)m * (DTR + 2 * DS) + r];
}

// Selective scan, one thread per (b, d_inner) channel; fused u*Dp and
// y *= silu(z) gate; writes y in bf16 for the output GEMM.
__global__ void scan_kernel(const __bf16* __restrict__ dt_bf,   // [BL,DI]
                            const __bf16* __restrict__ u_bf,    // [BL,DI]
                            const float*  __restrict__ dbl,     // [BL,64]
                            const __bf16* __restrict__ xz_bf,   // [BL,2*DI]
                            const float*  __restrict__ Alog_l,  // [DI,DS]
                            const float*  __restrict__ Dp_l,    // [DI]
                            __bf16* __restrict__ y_bf)          // [BL,DI]
{
    int tid = blockIdx.x * blockDim.x + threadIdx.x;
    if (tid >= BB * DI) return;
    int d = tid % DI;
    int b = tid / DI;

    float A[DS];
#pragma unroll
    for (int s = 0; s < DS; ++s) A[s] = -__expf(Alog_l[d * DS + s]);
    float Dd = Dp_l[d];
    float st[DS];
#pragma unroll
    for (int s = 0; s < DS; ++s) st[s] = 0.f;

    for (int t = 0; t < LL; ++t) {
        size_t row = (size_t)(b * LL + t);
        float dtv = (float)dt_bf[row * DI + d];
        float uv  = (float)u_bf[row * DI + d];
        const float* bc = dbl + row * (DTR + 2 * DS);
        float y = 0.f;
#pragma unroll
        for (int s = 0; s < DS; ++s) {
            float dA = __expf(dtv * A[s]);
            st[s] = dA * st[s] + dtv * bc[DTR + s] * uv;
            y += st[s] * bc[DTR + DS + s];
        }
        y += uv * Dd;
        float z  = (float)xz_bf[row * (2 * DI) + DI + d];
        float sz = z / (1.f + __expf(-z));
        y_bf[row * DI + d] = (__bf16)(y * sz);
    }
}

// mean over L
__global__ void pool_kernel(const float* __restrict__ h,
                            float* __restrict__ pooled)
{
    int idx = blockIdx.x * blockDim.x + threadIdx.x;
    if (idx >= BB * DM) return;
    int dm = idx % DM;
    int b  = idx / DM;
    const float* p = h + (size_t)b * LL * DM + dm;
    float s = 0.f;
    for (int t = 0; t < LL; ++t) s += p[(size_t)t * DM];
    pooled[idx] = s * (1.f / (float)LL);
}

// LayerNorm + classifier head; one block (512 threads) per batch element
__global__ void head_kernel(const float* __restrict__ pooled,
                            const float* __restrict__ nw,
                            const float* __restrict__ nb,
                            const float* __restrict__ clw,
                            const float* __restrict__ clb,
                            float* __restrict__ out)
{
    __shared__ float sh[DM];
    __shared__ float ln[DM];
    int b = blockIdx.x, t = threadIdx.x;
    float v = pooled[b * DM + t];
    sh[t] = v;
    __syncthreads();
    for (int s = DM / 2; s > 0; s >>= 1) {
        if (t < s) sh[t] += sh[t + s];
        __syncthreads();
    }
    float mu = sh[0] * (1.f / DM);
    __syncthreads();
    float dv = v - mu;
    sh[t] = dv * dv;
    __syncthreads();
    for (int s = DM / 2; s > 0; s >>= 1) {
        if (t < s) sh[t] += sh[t + s];
        __syncthreads();
    }
    float var = sh[0] * (1.f / DM);
    __syncthreads();
    ln[t] = dv * rsqrtf(var + 1e-5f) * nw[t] + nb[t];
    __syncthreads();
    if (t < NC) {
        float s = clb[t];
        for (int d = 0; d < DM; ++d) s += ln[d] * clw[t * DM + d];
        out[b * NC + t] = s;
    }
}

// ---------------------------------------------------------------------------
// Host-side orchestration
// ---------------------------------------------------------------------------
static inline void* ws_alloc(uint8_t* base, size_t& off, size_t bytes)
{
    off = (off + 255) & ~(size_t)255;
    void* p = base + off;
    off += bytes;
    return p;
}

extern "C" void kernel_launch(void* const* d_in, const int* in_sizes, int n_in,
                              void* d_out, int out_size, void* d_ws, size_t ws_size,
                              hipStream_t stream)
{
    (void)in_sizes; (void)n_in; (void)out_size; (void)ws_size;

    const float* x    = (const float*)d_in[0];
    const float* in_w = (const float*)d_in[1];
    const float* in_b = (const float*)d_in[2];
    const float* ipw  = (const float*)d_in[3];
    const float* cw   = (const float*)d_in[4];
    const float* cb   = (const float*)d_in[5];
    const float* xpw  = (const float*)d_in[6];
    const float* dtw  = (const float*)d_in[7];
    const float* dtb  = (const float*)d_in[8];
    const float* Alog = (const float*)d_in[9];
    const float* Dp   = (const float*)d_in[10];
    const float* opw  = (const float*)d_in[11];
    const float* nw   = (const float*)d_in[12];
    const float* nb   = (const float*)d_in[13];
    const float* clw  = (const float*)d_in[14];
    const float* clb  = (const float*)d_in[15];
    float* out        = (float*)d_out;

    uint8_t* ws = (uint8_t*)d_ws;
    size_t off = 0;
    float*  h      = (float*) ws_alloc(ws, off, (size_t)BL * DM * 4);
    __bf16* h_bf   = (__bf16*)ws_alloc(ws, off, (size_t)BL * DM * 2);
    __bf16* xz_bf  = (__bf16*)ws_alloc(ws, off, (size_t)BL * 2 * DI * 2);
    __bf16* u_bf   = (__bf16*)ws_alloc(ws, off, (size_t)BL * DI * 2);
    float*  dbl    = (float*) ws_alloc(ws, off, (size_t)BL * (DTR + 2 * DS) * 4);
    __bf16* dtA_bf = (__bf16*)ws_alloc(ws, off, (size_t)BL * DTR * 2);
    __bf16* dt_bf  = (__bf16*)ws_alloc(ws, off, (size_t)BL * DI * 2);
    __bf16* y_bf   = (__bf16*)ws_alloc(ws, off, (size_t)BL * DI * 2);
    float*  pooled = (float*) ws_alloc(ws, off, (size_t)BB * DM * 4);
    __bf16* ipw_bf = (__bf16*)ws_alloc(ws, off, (size_t)NL * 2 * DI * DM * 2);
    __bf16* xpw_bf = (__bf16*)ws_alloc(ws, off, (size_t)NL * (DTR + 2 * DS) * DI * 2);
    __bf16* dtw_bf = (__bf16*)ws_alloc(ws, off, (size_t)NL * DI * DTR * 2);
    __bf16* opw_bf = (__bf16*)ws_alloc(ws, off, (size_t)NL * DM * DI * 2);

    const int TPB = 256;
    auto blocks = [](long long n, int tpb) { return (unsigned)((n + tpb - 1) / tpb); };

    // ---- weight precision conversion (f32 -> bf16), once per launch ----
    { int n = NL * 2 * DI * DM;          cvt_f32_bf16<<<blocks(n, TPB), TPB, 0, stream>>>(ipw, ipw_bf, n); }
    { int n = NL * (DTR + 2 * DS) * DI;  cvt_f32_bf16<<<blocks(n, TPB), TPB, 0, stream>>>(xpw, xpw_bf, n); }
    { int n = NL * DI * DTR;             cvt_f32_bf16<<<blocks(n, TPB), TPB, 0, stream>>>(dtw, dtw_bf, n); }
    { int n = NL * DM * DI;              cvt_f32_bf16<<<blocks(n, TPB), TPB, 0, stream>>>(opw, opw_bf, n); }

    // ---- embedding ----
    embed_kernel<<<blocks((long long)BL * DM, TPB), TPB, 0, stream>>>(x, in_w, in_b, h, h_bf);

    for (int l = 0; l < NL; ++l) {
        const __bf16* ipw_l = ipw_bf + (size_t)l * 2 * DI * DM;
        const __bf16* xpw_l = xpw_bf + (size_t)l * (DTR + 2 * DS) * DI;
        const __bf16* dtw_l = dtw_bf + (size_t)l * DI * DTR;
        const __bf16* opw_l = opw_bf + (size_t)l * DM * DI;

        // xz = h @ ipw^T        (M=BL, N=2048, K=512) -> bf16
        gemm_bf16_wmma<<<dim3(BL / TM, (2 * DI + TN - 1) / TN), 256, 0, stream>>>(
            h_bf, DM, ipw_l, nullptr, xz_bf, 2 * DI, BL, 2 * DI, DM, GEMM_BF16);

        // depthwise causal conv + SiLU -> u (bf16)
        conv_silu_kernel<<<blocks((long long)BL * DI, TPB), TPB, 0, stream>>>(
            xz_bf, cw + (size_t)l * DI * DC, cb + (size_t)l * DI, u_bf);

        // dbl = u @ xpw^T       (M=BL, N=64, K=1024) -> f32
        gemm_bf16_wmma<<<dim3(BL / TM, (DTR + 2 * DS + TN - 1) / TN), 256, 0, stream>>>(
            u_bf, DI, xpw_l, nullptr, dbl, DTR + 2 * DS, BL, DTR + 2 * DS, DI, GEMM_F32);

        // bf16 copy of dt-rank slice
        extract_dtA_kernel<<<blocks((long long)BL * DTR, TPB), TPB, 0, stream>>>(dbl, dtA_bf);

        // dt = softplus(dtA @ dtw^T + dtb)   (M=BL, N=1024, K=32) -> bf16
        gemm_bf16_wmma<<<dim3(BL / TM, (DI + TN - 1) / TN), 256, 0, stream>>>(
            dtA_bf, DTR, dtw_l, dtb + (size_t)l * DI, dt_bf, DI, BL, DI, DTR,
            GEMM_SP_BIAS_BF16);

        // selective scan + D-skip + silu(z) gate -> y (bf16)
        scan_kernel<<<blocks((long long)BB * DI, TPB), TPB, 0, stream>>>(
            dt_bf, u_bf, dbl, xz_bf,
            Alog + (size_t)l * DI * DS, Dp + (size_t)l * DI, y_bf);

        // h = y @ opw^T         (M=BL, N=512, K=1024) -> f32
        gemm_bf16_wmma<<<dim3(BL / TM, (DM + TN - 1) / TN), 256, 0, stream>>>(
            y_bf, DI, opw_l, nullptr, h, DM, BL, DM, DI, GEMM_F32);

        // refresh bf16 copy of h for next layer's GEMM
        { int n = BL * DM; cvt_f32_bf16<<<blocks(n, TPB), TPB, 0, stream>>>(h, h_bf, n); }
    }

    // ---- mean pool + LayerNorm + classifier head ----
    pool_kernel<<<blocks((long long)BB * DM, TPB), TPB, 0, stream>>>(h, pooled);
    head_kernel<<<BB, DM, 0, stream>>>(pooled, nw, nb, clw, clb, out);
}
